// VShiftAttention_20134806684365
// MI455X (gfx1250) — matrix operand order repaired and checked by reading, not compiled, not run on hardware
//
#include <hip/hip_runtime.h>
#include <hip/hip_bf16.h>
#include <math.h>

// ---------------------------------------------------------------------------
// VShiftAttention for MI455X (gfx1250): f16 WMMA everywhere, flash attention
// with TDM (tensor_load_to_lds) double-buffered V staging.
// ---------------------------------------------------------------------------

typedef _Float16 half_t;
typedef __attribute__((ext_vector_type(16))) _Float16 v16h;
typedef __attribute__((ext_vector_type(8)))  _Float16 v8h;
typedef __attribute__((ext_vector_type(8)))  float    v8f;
typedef __attribute__((ext_vector_type(4)))  float    v4f;
typedef __attribute__((ext_vector_type(4)))  unsigned int v4u;
typedef __attribute__((ext_vector_type(8)))  int      v8i;
typedef __attribute__((ext_vector_type(4)))  int      v4i;

#define HH   16
#define DD   128
#define SS   2048
#define BB   2
#define EE   128
#define HD   2048          // H*D
#define MM   4096          // B*S
#define SCL  0.08838834764831845f   // 1/sqrt(128)

static __device__ __forceinline__ v8f wmma16(v16h a, v16h b, v8f c) {
  // D = A(16x32 f16) * B(32x16 f16) + C(16x16 f32)
  return __builtin_amdgcn_wmma_f32_16x16x32_f16(false, a, false, b,
                                                (short)0, c, false, false);
}

// A-fragment (16x32, lane = M row): per ISA 7.12.2 each lane holds two
// contiguous 8-half chunks at K = {8*lh, 8*lh+16}.
static __device__ __forceinline__ v16h load_fragA(const half_t* rowPtr, int kBase, int lh) {
  const half_t* p = rowPtr + kBase + lh * 8;
  v8h a = *(const v8h*)(p);
  v8h b = *(const v8h*)(p + 16);
  v16h r;
#pragma unroll
  for (int i = 0; i < 8; ++i) { r[i] = a[i]; r[i + 8] = b[i]; }
  return r;
}

// A-fragment sourced from fp32 rows (QKV GEMM input), converted to f16.
static __device__ __forceinline__ v16h load_fragA_f32(const float* rowPtr, int kBase, int lh) {
  const float* p = rowPtr + kBase + lh * 8;
  v4f a0 = *(const v4f*)(p);
  v4f a1 = *(const v4f*)(p + 4);
  v4f b0 = *(const v4f*)(p + 16);
  v4f b1 = *(const v4f*)(p + 20);
  v16h r;
#pragma unroll
  for (int i = 0; i < 4; ++i) {
    r[i]      = (half_t)a0[i];
    r[i + 4]  = (half_t)a1[i];
    r[i + 8]  = (half_t)b0[i];
    r[i + 12] = (half_t)b1[i];
  }
  return r;
}

// B-fragment (32x16, lane = N column): each lane holds two contiguous
// 8-half chunks at K = {8*lh, 8*lh+16}. rowPtr points at this lane's N-major data.
static __device__ __forceinline__ v16h load_fragB(const half_t* rowPtr, int kBase, int lh) {
  const half_t* p = rowPtr + kBase + lh * 8;
  v8h a = *(const v8h*)(p);
  v8h b = *(const v8h*)(p + 16);
  v16h r;
#pragma unroll
  for (int i = 0; i < 8; ++i) { r[i] = a[i]; r[i + 8] = b[i]; }
  return r;
}

// B-fragment gathered from a row-major LDS tile [32][128]: lane = column n,
// K (rows) strided by 128 halves.
static __device__ __forceinline__ v16h load_fragB_strided(const half_t* tile,
                                                          int ncol, int lh) {
  v16h r;
#pragma unroll
  for (int i = 0; i < 16; ++i) r[i] = tile[(size_t)(lh * 8 + i + (i & 8)) * DD + ncol];
  return r;
}

static __device__ __forceinline__ float rmax16(float x) {
#pragma unroll
  for (int m = 1; m < 16; m <<= 1) x = fmaxf(x, __shfl_xor(x, m, 32));
  return x;
}
static __device__ __forceinline__ float rsum16(float x) {
#pragma unroll
  for (int m = 1; m < 16; m <<= 1) x += __shfl_xor(x, m, 32);
  return x;
}

// ---------------------------------------------------------------------------
// TDM: async-load a 32x128 f16 tile (row-major, row stride 128 halves) from
// global memory into LDS.  D# per CDNA5 ISA.
// Tracked by TENSORcnt; consume after s_wait_tensorcnt.
// Toolchain is the 6-arg clang-23 variant:
//   (uint32x4 g0, int32x8 g1, int32x4 g2, int32x4 g3, int32x8 gx, i32 cpol)
// ---------------------------------------------------------------------------
static __device__ __forceinline__ void tdm_load_v_tile(const half_t* gsrc,
                                                       unsigned int lds_byte_off) {
  unsigned long long ga = (unsigned long long)(uintptr_t)gsrc;
  v4u g0;
  g0[0] = 1u;                                         // count=1, user descriptor
  g0[1] = lds_byte_off;                               // lds_addr (bytes)
  g0[2] = (unsigned int)(ga & 0xffffffffu);           // global_addr[31:0]
  g0[3] = (unsigned int)((ga >> 32) & 0x01ffffffu)    // global_addr[56:32]
        | 0x80000000u;                                // type = 2 ("image")
  v8i g1;
  g1[0] = (int)(1u << 16);        // workgroup_mask=0, data_size=1 (2 bytes)
  g1[1] = (int)(128u << 16);      // tensor_dim0[15:0]=128 at bits [63:48]
  g1[2] = (int)(2048u << 16);     // tensor_dim1[15:0]=2048 at bits [95:80]
  g1[3] = (int)(128u << 16);      // tile_dim0=128 at bits [127:112]
  g1[4] = 32;                     // tile_dim1=32 at bits [143:128]
  g1[5] = 128;                    // tensor_dim0_stride[31:0]=128
  g1[6] = 0;                      // dim0_stride[47:32]=0, dim1_stride lo=0
  g1[7] = 0;
  v4i z4 = {0, 0, 0, 0};          // groups 2/3 unused (2D tile)
  v8i z8 = {0, 0, 0, 0, 0, 0, 0, 0};
  __builtin_amdgcn_tensor_load_to_lds(g0, g1, z4, z4, z8, 0);
}

// ---------------------------------------------------------------------------
// Kernel 0: transpose + fp32->f16 convert a weight matrix. W:[K][N] -> WT:[N][K]
// ---------------------------------------------------------------------------
__global__ void vshift_transpose_w(const float* __restrict__ W,
                                   half_t* __restrict__ WT, int K, int N) {
  int idx = blockIdx.x * blockDim.x + threadIdx.x;
  if (idx >= K * N) return;
  int n = idx / K;
  int k = idx - n * K;
  WT[idx] = (half_t)W[(size_t)k * N + n];
}

// ---------------------------------------------------------------------------
// Kernel 1: fused QKV projection + RoPE.  One wave per 16x16 output tile.
//   which = 0:Q(rope) 1:K(rope) 2:V   -> f16 [B][H][S][D]
// ---------------------------------------------------------------------------
__global__ void vshift_qkv_rope(const float* __restrict__ X,    // [4096][128]
                                const half_t* __restrict__ WT,  // [3][2048][128]
                                half_t* __restrict__ qh,
                                half_t* __restrict__ kh,
                                half_t* __restrict__ vh) {
  int gw   = (blockIdx.x * blockDim.x + threadIdx.x) >> 5;
  int lane = threadIdx.x & 31;
  int lh   = lane >> 4;
  int ln   = lane & 15;

  int which = gw / (256 * 128);           // 0..2
  int rem   = gw - which * (256 * 128);
  int mt    = rem >> 7;                   // 0..255  (M tiles of 16)
  int nt    = rem & 127;                  // 0..127  (N tiles of 16)

  const float*  arow = X + (size_t)(mt * 16 + ln) * EE;
  const half_t* brow = WT + (size_t)which * HD * EE + (size_t)(nt * 16 + ln) * EE;

  v8f acc = {};
#pragma unroll
  for (int ks = 0; ks < 4; ++ks) {
    v16h a = load_fragA_f32(arow, ks * 32, lh);
    v16h b = load_fragB(brow, ks * 32, lh);
    acc = wmma16(a, b, acc);
  }

  half_t* outp = (which == 0) ? qh : (which == 1) ? kh : vh;
  int ncol = nt * 16 + ln;                 // column in [0,2048)
  int h    = ncol >> 7;
  int d    = ncol & 127;
  int b    = (mt * 16) >> 11;              // batch
  int sb   = (mt * 16) & 2047;             // sequence base of tile
  bool rope = (which < 2);

  float invf = __powf(10000.0f, -(float)(2 * (d >> 1)) / 128.0f);
  size_t obase = (((size_t)b * HH + h) * SS) * DD + d;

#pragma unroll
  for (int v = 0; v < 8; ++v) {
    int   sv  = sb + v + 8 * lh;           // sequence position of this element
    float val = acc[v];
    float part = __shfl_xor(val, 1, 32);   // partner column d^1
    if (rope) {
      float ang = (float)sv * invf;
      float sn, cs;
      __sincosf(ang, &sn, &cs);
      val = (d & 1) ? (part * sn + val * cs) : (val * cs - part * sn);
    }
    outp[obase + (size_t)sv * DD] = (half_t)val;
  }
}

// ---------------------------------------------------------------------------
// Kernel 2: flash attention, one wave per (b,h,16-row query tile), 2 waves/blk.
// V tiles are staged global->LDS with the Tensor Data Mover, double-buffered
// so tile i+1 streams in while tile i is consumed.  Softmax denominator
// includes the diagonal; diagonal zeroed in P (matches reference ordering).
// ---------------------------------------------------------------------------
__global__ void vshift_flash_attn(const half_t* __restrict__ qh,
                                  const half_t* __restrict__ kh,
                                  const half_t* __restrict__ vh,
                                  half_t* __restrict__ attnO) {  // [B][S][H*D] f16
  __shared__ half_t ldsV[2][2][32 * 128];  // [wave][buffer] V tile (8 KB each)
  __shared__ half_t ldsP[2][16 * 32];      // [wave] P tile (1 KB each)

  int gw   = (blockIdx.x * blockDim.x + threadIdx.x) >> 5;
  int wv   = (threadIdx.x >> 5) & 1;
  int lane = threadIdx.x & 31;
  int lh   = lane >> 4;
  int ln   = lane & 15;

  int qt = gw & 127;                       // query tile within (b,h)
  int bh = gw >> 7;
  int h  = bh & 15;
  int b  = bh >> 4;
  int q0 = qt * 16;

  const half_t* Q = qh + (((size_t)b * HH + h) * SS) * DD;
  const half_t* K = kh + (((size_t)b * HH + h) * SS) * DD;
  const half_t* V = vh + (((size_t)b * HH + h) * SS) * DD;

  unsigned int vbuf_off[2];
  vbuf_off[0] = (unsigned int)(uintptr_t)(void*)&ldsV[wv][0][0];
  vbuf_off[1] = (unsigned int)(uintptr_t)(void*)&ldsV[wv][1][0];

  // Q fragments, held in VGPRs for the whole key loop.
  v16h qf[4];
  const half_t* qrow = Q + (size_t)(q0 + ln) * DD;
#pragma unroll
  for (int ks = 0; ks < 4; ++ks) qf[ks] = load_fragA(qrow, ks * 32, lh);

  float m_i[8], l_i[8];
  v8f   oacc[8];
#pragma unroll
  for (int v = 0; v < 8; ++v) { m_i[v] = -1e30f; l_i[v] = 0.0f; }
#pragma unroll
  for (int n = 0; n < 8; ++n) oacc[n] = (v8f){};

  int tEnd   = q0 + 16;                    // exclusive causal bound
  int nTiles = (tEnd + 31) >> 5;           // 32-key tiles

  // Kick off TDM for the first V tile.
  tdm_load_v_tile(V, vbuf_off[0]);

  for (int ti = 0; ti < nTiles; ++ti) {
    int t0 = ti * 32;

    // Issue TDM for the next V tile into the other buffer (pipelined).
    if (ti + 1 < nTiles)
      tdm_load_v_tile(V + (size_t)(t0 + 32) * DD, vbuf_off[(ti + 1) & 1]);

    // Prefetch next K tile rows -> global_prefetch_b8
    __builtin_prefetch(K + (size_t)(t0 + 32 + lane) * DD, 0, 1);

    // ---- scores: two 16-column halves of the 32-key tile -------------------
    v8f sc[2] = { (v8f){}, (v8f){} };
#pragma unroll
    for (int half = 0; half < 2; ++half) {
      const half_t* krow = K + (size_t)(t0 + half * 16 + ln) * DD;
#pragma unroll
      for (int ks = 0; ks < 4; ++ks) {
        v16h kf = load_fragB(krow, ks * 32, lh);   // lane = key column t
        sc[half] = wmma16(qf[ks], kf, sc[half]);
      }
    }

    // ---- online softmax with causal mask ----------------------------------
    int tc0 = t0 + ln;
    int tc1 = t0 + 16 + ln;
#pragma unroll
    for (int v = 0; v < 8; ++v) {
      int   srow = q0 + v + 8 * lh;
      float x0 = sc[0][v] * SCL;
      float x1 = sc[1][v] * SCL;
      if (tc0 > srow) x0 = -1e30f;
      if (tc1 > srow) x1 = -1e30f;
      float mx = rmax16(fmaxf(x0, x1));
      float mn = fmaxf(m_i[v], mx);
      float alpha = __expf(m_i[v] - mn);
      float p0 = __expf(x0 - mn);
      float p1 = __expf(x1 - mn);
      float rs = rsum16(p0 + p1);          // includes diagonal term
      l_i[v] = l_i[v] * alpha + rs;
      m_i[v] = mn;
      if (tc0 == srow) p0 = 0.0f;          // drop diagonal from P
      if (tc1 == srow) p1 = 0.0f;
#pragma unroll
      for (int n = 0; n < 8; ++n) oacc[n][v] *= alpha;
      half_t* prow = &ldsP[wv][(v + 8 * lh) * 32];
      prow[ln]      = (half_t)p0;
      prow[16 + ln] = (half_t)p1;
    }

    // P stores -> P loads cross-lane hazard within the wave: drain DScnt.
    asm volatile("s_wait_dscnt 0x0" ::: "memory");
    // Current V tile must have landed; the next tile may still be in flight.
    if (ti + 1 < nTiles) __builtin_amdgcn_s_wait_tensorcnt(1);
    else                 __builtin_amdgcn_s_wait_tensorcnt(0);

    // ---- P @ V : A = P (16x32 over keys), B from row-major LDS V tile -----
    v16h pf = load_fragA(&ldsP[wv][ln * 32], 0, lh);
    const half_t* vtile = &ldsV[wv][ti & 1][0];
#pragma unroll
    for (int n = 0; n < 8; ++n) {
      v16h vf = load_fragB_strided(vtile, n * 16 + ln, lh);
      oacc[n] = wmma16(pf, vf, oacc[n]);
    }
  }

  // ---- epilogue: normalize (denominator includes diagonal) and store f16 --
#pragma unroll
  for (int n = 0; n < 8; ++n) {
    int dcol = n * 16 + ln;
#pragma unroll
    for (int v = 0; v < 8; ++v) {
      int srow = q0 + v + 8 * lh;
      float o = oacc[n][v] / l_i[v];
      attnO[((size_t)b * SS + srow) * HD + (size_t)h * DD + dcol] = (half_t)o;
    }
  }
}

// ---------------------------------------------------------------------------
// Kernel 3: output projection  out = attnO(4096x2048 f16) @ Wo(2048x128)
// ---------------------------------------------------------------------------
__global__ void vshift_out_proj(const half_t* __restrict__ A,    // [4096][2048]
                                const half_t* __restrict__ WoT,  // [128][2048]
                                float* __restrict__ out) {       // [4096][128]
  int gw   = (blockIdx.x * blockDim.x + threadIdx.x) >> 5;
  int lane = threadIdx.x & 31;
  int lh   = lane >> 4;
  int ln   = lane & 15;
  int mt   = gw >> 3;                      // 0..255
  int nt   = gw & 7;                       // 0..7

  const half_t* arow = A   + (size_t)(mt * 16 + ln) * HD;
  const half_t* brow = WoT + (size_t)(nt * 16 + ln) * HD;

  v8f acc = {};
  for (int ks = 0; ks < 64; ++ks) {
    __builtin_prefetch(arow + ks * 32 + 512, 0, 1);
    v16h a = load_fragA(arow, ks * 32, lh);
    v16h b = load_fragB(brow, ks * 32, lh);
    acc = wmma16(a, b, acc);
  }
#pragma unroll
  for (int v = 0; v < 8; ++v)
    out[(size_t)(mt * 16 + v + 8 * lh) * EE + nt * 16 + ln] = acc[v];
}

// ---------------------------------------------------------------------------
// Host-side launcher. Workspace layout (halves):
//  WqT|WkT|WvT (3 x 2048*128) | WoT (128*2048) | qh|kh|vh (3 x 2*16*2048*128)
//  | attnO (2*2048*2048)   ->  ~69 MB total.
// ---------------------------------------------------------------------------
extern "C" void kernel_launch(void* const* d_in, const int* in_sizes, int n_in,
                              void* d_out, int out_size, void* d_ws, size_t ws_size,
                              hipStream_t stream) {
  const float* q  = (const float*)d_in[0];
  const float* Wq = (const float*)d_in[1];
  const float* Wk = (const float*)d_in[2];
  const float* Wv = (const float*)d_in[3];
  const float* Wo = (const float*)d_in[4];

  half_t* WqT = (half_t*)d_ws;
  half_t* WkT = WqT + (size_t)HD * EE;
  half_t* WvT = WkT + (size_t)HD * EE;
  half_t* WoT = WvT + (size_t)HD * EE;
  half_t* qh  = WoT + (size_t)EE * HD;
  half_t* kh  = qh  + (size_t)BB * HH * SS * DD;
  half_t* vh  = kh  + (size_t)BB * HH * SS * DD;
  half_t* aO  = vh  + (size_t)BB * HH * SS * DD;

  // 0) weight transpose + f16 convert
  vshift_transpose_w<<<(HD * EE + 255) / 256, 256, 0, stream>>>(Wq, WqT, EE, HD);
  vshift_transpose_w<<<(HD * EE + 255) / 256, 256, 0, stream>>>(Wk, WkT, EE, HD);
  vshift_transpose_w<<<(HD * EE + 255) / 256, 256, 0, stream>>>(Wv, WvT, EE, HD);
  vshift_transpose_w<<<(EE * HD + 255) / 256, 256, 0, stream>>>(Wo, WoT, HD, EE);

  // 1) QKV projection + RoPE: 3 * 256 * 128 waves, 4 waves/block
  vshift_qkv_rope<<<(3 * 256 * 128) / 4, 128, 0, stream>>>(q, WqT, qh, kh, vh);

  // 2) flash attention: B*H*(S/16) = 4096 waves, 2 waves/block (TDM buffers)
  vshift_flash_attn<<<(BB * HH * (SS / 16)) / 2, 64, 0, stream>>>(qh, kh, vh, aO);

  // 3) output projection: 256*8 = 2048 waves, 4 waves/block
  vshift_out_proj<<<(256 * 8) / 4, 128, 0, stream>>>(aO, WoT, (float*)d_out);
}
